// SGFormerEncoderLayer_46437186404988
// MI455X (gfx1250) — matrix operand order repaired and checked by reading, not compiled, not run on hardware
//
#include <hip/hip_runtime.h>
#include <hip/hip_bf16.h>
#include <math.h>

// SGFormer encoder layer for gfx1250 (MI455X), bf16 WMMA everywhere.
#define NN 4096
#define CC 512
#define HH 8
#define DD 64
#define EE 131072
#define C3 1536

typedef unsigned short u16;
typedef __bf16 v16bf __attribute__((ext_vector_type(16)));
typedef float  f32x8 __attribute__((ext_vector_type(8)));

union FragU { v16bf v; uint4 q[2]; };

__device__ __forceinline__ u16 f2bf(float f) {
  union { float f; unsigned u; } c; c.f = f;
  unsigned u = c.u;
  return (u16)((u + 0x7FFFu + ((u >> 16) & 1u)) >> 16);  // RNE to bf16
}

// ---- DPP row-xor reductions within the 16-lane row that owns a C/D-layout row ----
template <int MASK>
__device__ __forceinline__ float xor16(float x) {
  // DPP16 ROW_XMASK: ctrl = 0x160 | mask (xor within a row of 16 lanes)
  return __int_as_float(
      __builtin_amdgcn_update_dpp(0, __float_as_int(x), 0x160 + MASK, 0xf, 0xf, true));
}
__device__ __forceinline__ float redmax16(float x) {
  x = fmaxf(x, xor16<1>(x));
  x = fmaxf(x, xor16<2>(x));
  x = fmaxf(x, xor16<4>(x));
  x = fmaxf(x, xor16<8>(x));
  return x;
}
__device__ __forceinline__ float redsum16(float x) {
  x += xor16<1>(x);
  x += xor16<2>(x);
  x += xor16<4>(x);
  x += xor16<8>(x);
  return x;
}

// ---- WMMA fragment loads (wave32 layouts per CDNA5 ISA 7.12.2) ----
// A (16x32 bf16): lane m=lane&15 holds row m; K chunks [hf*8,+8) and [16+hf*8,+8)
__device__ __forceinline__ v16bf load_a_frag(const u16* __restrict__ base, int lda,
                                             int row0, int k0, int lane) {
  const int hf = lane >> 4, m = lane & 15;
  const u16* p = base + (size_t)(row0 + m) * lda + k0 + hf * 8;
  FragU f;
  f.q[0] = *(const uint4*)p;
  f.q[1] = *(const uint4*)(p + 16);
  return f.v;
}
// B (32x16 bf16) for A@W^T: lane n=lane&15 holds column n (= row n0+n of W);
// K chunks [hf*8,+8) and [16+hf*8,+8)
__device__ __forceinline__ v16bf load_bt_frag(const u16* __restrict__ W, int ldw,
                                              int n0, int k0, int lane) {
  const int hf = lane >> 4, n = lane & 15;
  const u16* p = W + (size_t)(n0 + n) * ldw + k0 + hf * 8;
  FragU f;
  f.q[0] = *(const uint4*)p;
  f.q[1] = *(const uint4*)(p + 16);
  return f.v;
}

__device__ __forceinline__ float gelu_exact(float x) {
  return 0.5f * x * (1.0f + erff(x * 0.70710678118654752f));
}

// ---- elementwise helpers (128-bit vectorized) ----
__global__ void cvt_bf16_kernel(const float* __restrict__ in, u16* __restrict__ out, int n4) {
  int i = blockIdx.x * 256 + threadIdx.x;  // over n/4
  if (i < n4) {
    float4 f = ((const float4*)in)[i];
    uint2 o;
    o.x = (unsigned)f2bf(f.x) | ((unsigned)f2bf(f.y) << 16);
    o.y = (unsigned)f2bf(f.z) | ((unsigned)f2bf(f.w) << 16);
    ((uint2*)out)[i] = o;
  }
}
__global__ void zero_kernel(float4* __restrict__ p, int n4) {
  int i = blockIdx.x * 256 + threadIdx.x;
  if (i < n4) p[i] = make_float4(0.f, 0.f, 0.f, 0.f);
}

// ---- generic OUT = A[M,K](bf16) @ W[Nout,K]^T (bf16) + bias, templated epilogue ----
// EPI: 0 = f32 store, 1 = bf16 store, 2 = exact GELU -> bf16 store
template <int EPI>
__global__ __launch_bounds__(256) void gemm_xwT(const u16* __restrict__ A,
                                                const u16* __restrict__ W,
                                                const float* __restrict__ bias,
                                                float* __restrict__ outF,
                                                u16* __restrict__ outB,
                                                int K, int ldo) {
  const int lane = threadIdx.x & 31;
  const int wave = threadIdx.x >> 5;
  const int row0 = blockIdx.y * 128 + wave * 16;  // 8 waves -> 128 rows / block
  const int col0 = blockIdx.x * 64;               // 64 output cols / block
  f32x8 acc[4] = {};
  for (int k0 = 0; k0 < K; k0 += 32) {
    v16bf a = load_a_frag(A, K, row0, k0, lane);
#pragma unroll
    for (int j = 0; j < 4; ++j) {
      v16bf b = load_bt_frag(W, K, col0 + j * 16, k0, lane);
      acc[j] = __builtin_amdgcn_wmma_f32_16x16x32_bf16(false, a, false, b,
                                                       (short)0, acc[j], false, false);
    }
  }
  const int hf = lane >> 4, n = lane & 15;
#pragma unroll
  for (int j = 0; j < 4; ++j) {
    const int col = col0 + j * 16 + n;
    const float bb = bias[col];
#pragma unroll
    for (int r = 0; r < 8; ++r) {
      const int row = row0 + r + 8 * hf;
      float v = acc[j][r] + bb;
      if (EPI == 2) v = gelu_exact(v);
      if (EPI == 0) outF[(size_t)row * ldo + col] = v;
      else          outB[(size_t)row * ldo + col] = f2bf(v);
    }
  }
}

// ---- v-section transpose via LDS tiles: vt[c][n] = qkv[n][1024+c] ----
// 64x64 tile per block; coalesced 128-bit loads AND stores.
__global__ __launch_bounds__(256) void transpose_v_kernel(const u16* __restrict__ qkv,
                                                          u16* __restrict__ vt) {
  __shared__ __align__(16) u16 T[64][72];  // padded row (144B) to spread banks
  const int n0 = blockIdx.x * 64;          // node tile origin
  const int c0 = blockIdx.y * 64;          // channel tile origin
  const int t  = threadIdx.x;
  const int rr = t >> 3;                   // 0..31
  const int c8 = (t & 7) * 8;              // 0,8,...,56
#pragma unroll
  for (int i = 0; i < 2; ++i) {
    const int r = rr + i * 32;             // node row within tile
    const uint4 d = *(const uint4*)(qkv + (size_t)(n0 + r) * C3 + 1024 + c0 + c8);
    *(uint4*)(&T[r][c8]) = d;
  }
  __syncthreads();
#pragma unroll
  for (int i = 0; i < 2; ++i) {
    const int c = rr + i * 32;             // channel row of vt
    uint4 o;
    u16* po = (u16*)&o;
#pragma unroll
    for (int j = 0; j < 8; ++j) po[j] = T[c8 + j][c];
    *(uint4*)(vt + (size_t)(c0 + c) * NN + n0 + c8) = o;
  }
}

// ---- flash attention: one wave per (head, 16-query block), 64-key chunks ----
__global__ __launch_bounds__(256) void attn_kernel(const u16* __restrict__ qkv,
                                                   const u16* __restrict__ vt,
                                                   u16* __restrict__ ctx) {
  const int lane = threadIdx.x & 31;
  const int wave = threadIdx.x >> 5;
  const int gw = blockIdx.x * 8 + wave;   // 2048 waves
  const int h  = gw >> 8;                 // head
  const int r0 = (gw & 255) * 16;         // query block
  const int hf = lane >> 4, nn = lane & 15;

  __shared__ __align__(16) u16 Pl[8][16 * 64];
  u16* P = Pl[wave];

  v16bf qf[2];
#pragma unroll
  for (int kk = 0; kk < 2; ++kk)
    qf[kk] = load_a_frag(qkv, C3, r0, h * 64 + kk * 32, lane);

  f32x8 o[4] = {};
  float m[8], l[8];
#pragma unroll
  for (int r = 0; r < 8; ++r) { m[r] = -1e30f; l[r] = 0.0f; }

  const u16* kbase = qkv + CC;              // k section at columns [512,1024)
  const float SC = 0.18033688011112042f;    // (1/sqrt(64)) * log2(e)

  for (int kc = 0; kc < NN; kc += 64) {
    // prefetch next chunk's K/V lines into cache (global_prefetch_b8)
    {
      const int kn = kc + 64;
      __builtin_prefetch(kbase + (size_t)(kn + nn + hf * 16) * C3 + h * 64, 0, 3);
      __builtin_prefetch(vt + (size_t)(h * 64 + nn + hf * 16) * NN + kn, 0, 3);
    }
    // S tiles: 16 queries x 64 keys (8 WMMAs)
    f32x8 s[4] = {};
#pragma unroll
    for (int jn = 0; jn < 4; ++jn)
#pragma unroll
      for (int kk = 0; kk < 2; ++kk) {
        v16bf kb = load_bt_frag(kbase, C3, kc + jn * 16, h * 64 + kk * 32, lane);
        s[jn] = __builtin_amdgcn_wmma_f32_16x16x32_bf16(false, qf[kk], false, kb,
                                                        (short)0, s[jn], false, false);
      }
    // online softmax per row; raw logits, scale folded into exp2 argument
#pragma unroll
    for (int r = 0; r < 8; ++r) {
      const float a0 = s[0][r], a1 = s[1][r], a2 = s[2][r], a3 = s[3][r];
      float rm = redmax16(fmaxf(fmaxf(a0, a1), fmaxf(a2, a3)));
      const float mn = fmaxf(m[r], rm);
      const float al = __builtin_amdgcn_exp2f((m[r] - mn) * SC);
      m[r] = mn;
      const float p0 = __builtin_amdgcn_exp2f((a0 - mn) * SC);
      const float p1 = __builtin_amdgcn_exp2f((a1 - mn) * SC);
      const float p2 = __builtin_amdgcn_exp2f((a2 - mn) * SC);
      const float p3 = __builtin_amdgcn_exp2f((a3 - mn) * SC);
      const float rs = redsum16((p0 + p1) + (p2 + p3));
      l[r] = l[r] * al + rs;
      o[0][r] *= al; o[1][r] *= al; o[2][r] *= al; o[3][r] *= al;
      const int row = r + 8 * hf;
      P[row * 64 + nn]      = f2bf(p0);
      P[row * 64 + 16 + nn] = f2bf(p1);
      P[row * 64 + 32 + nn] = f2bf(p2);
      P[row * 64 + 48 + nn] = f2bf(p3);
    }
    // D-layout -> A-layout relayout through LDS (same wave: DS in-order, just wait)
    asm volatile("s_wait_dscnt 0" ::: "memory");
    FragU pf[2];
#pragma unroll
    for (int kk = 0; kk < 2; ++kk) {
      const u16* pp = P + nn * 64 + kk * 32 + hf * 8;
      pf[kk].q[0] = *(const uint4*)pp;
      pf[kk].q[1] = *(const uint4*)(pp + 16);
    }
    // O += P @ V  (V^T layout: vt[h*64+d][n], so B columns are contiguous in keys)
#pragma unroll
    for (int jd = 0; jd < 4; ++jd)
#pragma unroll
      for (int kk = 0; kk < 2; ++kk) {
        FragU vb;
        const u16* vp = vt + (size_t)(h * 64 + jd * 16 + nn) * NN + kc + kk * 32 + hf * 8;
        vb.q[0] = *(const uint4*)vp;
        vb.q[1] = *(const uint4*)(vp + 16);
        o[jd] = __builtin_amdgcn_wmma_f32_16x16x32_bf16(false, pf[kk].v, false, vb.v,
                                                        (short)0, o[jd], false, false);
      }
  }
  // normalize and store ctx (bf16, row-major [N,C])
#pragma unroll
  for (int jd = 0; jd < 4; ++jd)
#pragma unroll
    for (int r = 0; r < 8; ++r) {
      const int row = r0 + r + 8 * hf;
      const int col = h * 64 + jd * 16 + nn;
      ctx[(size_t)row * CC + col] = f2bf(o[jd][r] / l[r]);
    }
}

// ---- scatter-mean: one block per edge ----
__global__ void scatter_kernel(const int* __restrict__ src, const int* __restrict__ dst,
                               const float* __restrict__ lh, float* __restrict__ summed,
                               float* __restrict__ cnt) {
  const int e = blockIdx.x;
  const int s = src[e], d = dst[e];
  if (threadIdx.x == 0) atomicAdd(&cnt[d], 1.0f);
  for (int c = threadIdx.x; c < CC; c += 128)
    atomicAdd(&summed[(size_t)d * CC + c], lh[(size_t)s * CC + c]);
}

// ---- mix: 0.5*local_mean + 0.5*global, to bf16 (vectorized) ----
__global__ void mix_kernel(const float* __restrict__ summed, const float* __restrict__ cnt,
                           const float* __restrict__ gout, u16* __restrict__ mixedB) {
  int i = blockIdx.x * 256 + threadIdx.x;  // over N*C/4
  int n = i >> 7;                          // (i*4)/512
  float inv = 0.5f / fmaxf(cnt[n], 1.0f);
  float4 s = ((const float4*)summed)[i];
  float4 g = ((const float4*)gout)[i];
  uint2 o;
  o.x = (unsigned)f2bf(s.x * inv + 0.5f * g.x) | ((unsigned)f2bf(s.y * inv + 0.5f * g.y) << 16);
  o.y = (unsigned)f2bf(s.z * inv + 0.5f * g.z) | ((unsigned)f2bf(s.w * inv + 0.5f * g.w) << 16);
  ((uint2*)mixedB)[i] = o;
}

// ---- layernorm over C=512: out = LN(a+b)*g+be ; optional bf16 copy ----
__global__ __launch_bounds__(256) void ln512_kernel(const float* __restrict__ A,
                                                    const float* __restrict__ B,
                                                    const float* __restrict__ g,
                                                    const float* __restrict__ be,
                                                    float* __restrict__ outF,
                                                    u16* __restrict__ outB) {
  const int row = blockIdx.x, t = threadIdx.x;
  const float* a = A + (size_t)row * CC;
  const float* b = B + (size_t)row * CC;
  float x0 = a[t] + b[t], x1 = a[t + 256] + b[t + 256];
  __shared__ float sm[256];
  sm[t] = x0 + x1;
  __syncthreads();
  for (int st = 128; st > 0; st >>= 1) {
    if (t < st) sm[t] += sm[t + st];
    __syncthreads();
  }
  const float mu = sm[0] * (1.0f / 512.0f);
  __syncthreads();
  const float d0 = x0 - mu, d1 = x1 - mu;
  sm[t] = d0 * d0 + d1 * d1;
  __syncthreads();
  for (int st = 128; st > 0; st >>= 1) {
    if (t < st) sm[t] += sm[t + st];
    __syncthreads();
  }
  const float rstd = rsqrtf(sm[0] * (1.0f / 512.0f) + 1e-5f);
  const float y0 = d0 * rstd * g[t] + be[t];
  const float y1 = d1 * rstd * g[t + 256] + be[t + 256];
  outF[(size_t)row * CC + t]       = y0;
  outF[(size_t)row * CC + t + 256] = y1;
  if (outB) {
    outB[(size_t)row * CC + t]       = f2bf(y0);
    outB[(size_t)row * CC + t + 256] = f2bf(y1);
  }
}

extern "C" void kernel_launch(void* const* d_in, const int* in_sizes, int n_in,
                              void* d_out, int out_size, void* d_ws, size_t ws_size,
                              hipStream_t stream) {
  const float* x          = (const float*)d_in[0];
  const int*   ei         = (const int*)d_in[1];
  const float* local_w    = (const float*)d_in[2];
  const float* local_b    = (const float*)d_in[3];
  const float* in_proj_w  = (const float*)d_in[4];
  const float* in_proj_b  = (const float*)d_in[5];
  const float* attn_out_w = (const float*)d_in[6];
  const float* attn_out_b = (const float*)d_in[7];
  const float* output_w   = (const float*)d_in[8];
  const float* output_b   = (const float*)d_in[9];
  const float* norm1_g    = (const float*)d_in[10];
  const float* norm1_b    = (const float*)d_in[11];
  const float* norm2_g    = (const float*)d_in[12];
  const float* norm2_b    = (const float*)d_in[13];
  const float* ffn_w1     = (const float*)d_in[14];
  const float* ffn_b1     = (const float*)d_in[15];
  const float* ffn_w2     = (const float*)d_in[16];
  const float* ffn_b2     = (const float*)d_in[17];
  float* out = (float*)d_out;

  const int* src = ei;
  const int* dst = ei + EE;

  // ---- carve workspace (256B aligned regions) ----
  char* ws = (char*)d_ws;
  size_t off = 0;
  auto carve = [&](size_t bytes) -> void* {
    void* p = ws + off;
    off += (bytes + 255) & ~(size_t)255;
    return p;
  };
  u16*   xb      = (u16*)carve((size_t)NN * CC * 2);
  u16*   lwb     = (u16*)carve((size_t)CC * CC * 2);
  u16*   ipw     = (u16*)carve((size_t)C3 * CC * 2);
  u16*   aow     = (u16*)carve((size_t)CC * CC * 2);
  u16*   oww     = (u16*)carve((size_t)CC * CC * 2);
  u16*   f1w     = (u16*)carve((size_t)2 * CC * CC * 2);
  u16*   f2w     = (u16*)carve((size_t)2 * CC * CC * 2);
  float* local_h = (float*)carve((size_t)NN * CC * 4);
  float* summed  = (float*)carve((size_t)NN * CC * 4);
  float* cnt     = (float*)carve((size_t)NN * 4);
  u16*   qkvB    = (u16*)carve((size_t)NN * C3 * 2);
  u16*   vtb     = (u16*)carve((size_t)CC * NN * 2);
  u16*   ctxB    = (u16*)carve((size_t)NN * CC * 2);
  float* gout    = (float*)carve((size_t)NN * CC * 4);
  // reuse (lifetimes disjoint):
  float* proj    = local_h;        // free after scatter
  float* hiddenF = (float*)qkvB;   // free after attention
  u16*   hiddenB = vtb;            // free after attention
  u16*   mixedB  = ctxB;           // free after attn_out gemm
  u16*   ffB     = (u16*)summed;   // free after mix
  float* ff2     = gout;           // free after mix
  (void)ws_size; (void)n_in; (void)in_sizes; (void)out_size;

  auto cvt = [&](const float* s, u16* d, int n) {
    int n4 = n / 4;
    cvt_bf16_kernel<<<(n4 + 255) / 256, 256, 0, stream>>>(s, d, n4);
  };

  // 1) bf16 conversions
  cvt(x, xb, NN * CC);
  cvt(local_w, lwb, CC * CC);
  cvt(in_proj_w, ipw, C3 * CC);
  cvt(attn_out_w, aow, CC * CC);
  cvt(output_w, oww, CC * CC);
  cvt(ffn_w1, f1w, 2 * CC * CC);
  cvt(ffn_w2, f2w, 2 * CC * CC);

  // 2) local_h = x @ local_w^T + local_b  (f32)
  gemm_xwT<0><<<dim3(CC / 64, NN / 128), 256, 0, stream>>>(xb, lwb, local_b,
                                                           local_h, nullptr, CC, CC);
  // 3) zero scatter accumulators
  zero_kernel<<<(NN * CC / 4 + 255) / 256, 256, 0, stream>>>((float4*)summed, NN * CC / 4);
  zero_kernel<<<(NN / 4 + 255) / 256, 256, 0, stream>>>((float4*)cnt, NN / 4);
  // 4) scatter-mean numerators/counts
  scatter_kernel<<<EE, 128, 0, stream>>>(src, dst, local_h, summed, cnt);

  // 5) qkv = x @ in_proj^T + b  (bf16, [N,1536])
  gemm_xwT<1><<<dim3(C3 / 64, NN / 128), 256, 0, stream>>>(xb, ipw, in_proj_b,
                                                           nullptr, qkvB, CC, C3);
  // 6) v transpose for attention B-fragments (LDS-tiled, coalesced)
  transpose_v_kernel<<<dim3(NN / 64, CC / 64), 256, 0, stream>>>(qkvB, vtb);
  // 7) flash attention -> ctx (bf16)
  attn_kernel<<<(HH * (NN / 16)) / 8, 256, 0, stream>>>(qkvB, vtb, ctxB);
  // 8) global_out = ctx @ attn_out^T + b  (f32)
  gemm_xwT<0><<<dim3(CC / 64, NN / 128), 256, 0, stream>>>(ctxB, aow, attn_out_b,
                                                           gout, nullptr, CC, CC);
  // 9) mixed = 0.5*local_mean + 0.5*global  (bf16)
  mix_kernel<<<(NN * CC / 4) / 256, 256, 0, stream>>>(summed, cnt, gout, mixedB);
  // 10) proj = mixed @ output_w^T + b  (f32)
  gemm_xwT<0><<<dim3(CC / 64, NN / 128), 256, 0, stream>>>(mixedB, oww, output_b,
                                                           proj, nullptr, CC, CC);
  // 11) hidden = LN(x + proj)
  ln512_kernel<<<NN, 256, 0, stream>>>(x, proj, norm1_g, norm1_b, hiddenF, hiddenB);
  // 12) ff1 = gelu(hidden @ ffn_w1^T + b1)  (bf16, [N,1024])
  gemm_xwT<2><<<dim3((2 * CC) / 64, NN / 128), 256, 0, stream>>>(hiddenB, f1w, ffn_b1,
                                                                 nullptr, ffB, CC, 2 * CC);
  // 13) ff2 = ff1 @ ffn_w2^T + b2  (f32)
  gemm_xwT<0><<<dim3(CC / 64, NN / 128), 256, 0, stream>>>(ffB, f2w, ffn_b2,
                                                           ff2, nullptr, 2 * CC, CC);
  // 14) out = LN(hidden + ff2)
  ln512_kernel<<<NN, 256, 0, stream>>>(hiddenF, ff2, norm2_g, norm2_b, out, nullptr);
}
